// EncoderLSTMWithGAT_74217034875295
// MI455X (gfx1250) — matrix operand order repaired and checked by reading, not compiled, not run on hardware
//
// EncoderLSTMWithGAT for MI455X (gfx1250, wave32).
// All heavy GEMMs run on V_WMMA_F32_16X16X32_F16 (f16 inputs, fp32 accum).
// LSTM input projections are hoisted out of the time loop; only h@Whh stays
// sequential (fp32 VALU). Softmax/LN/gates are fp32 VALU, wave-level shuffles.
// Workspace requirement: ~620 MB (GAT intermediates dominate).
#include <hip/hip_runtime.h>
#include <hip/hip_bf16.h>

typedef _Float16 v8h  __attribute__((ext_vector_type(8)));
typedef _Float16 v16h __attribute__((ext_vector_type(16)));
typedef float    v8f  __attribute__((ext_vector_type(8)));

#define DI __device__ __forceinline__
DI float sigf(float x){ return 1.f/(1.f+expf(-x)); }

constexpr int BS=16, T=48, Nn=147, INs=294, Hh=128, AH=8, OUTs=294;
constexpr int BT=BS*T;        // 768
constexpr int INP=320;        // padded 294 -> %32
constexpr int NP=160;         // padded 147 -> %32 (and %16 rows)
constexpr int G1N=4*INs;      // 1176
constexpr int G1NP=1184;      // padded
constexpr int NH=Nn*Hh;       // 18816 (%32 == 0)

// ---------------------------------------------------------------------------
// Generic WMMA GEMM: C[m,n] = act( sum_k A[m,k]*B[n,k] + bias1[n] + bias2[n] )
// A: f16 [>=mt*16, lda], B: f16 [>=nt*16, ldb], K multiple of 32 (padded, zeros).
// Optional outputs: C (f32, row stride ldc), Ch (f16, ldch), ChT (f16,
// transposed, ChT[n*ldcht+m]).  Batch y: A+=y*aB, B+=y*bB,
// outputs += (y/bdiv)*oB1 + (y%bdiv)*oB2.   act: 0 none, 1 ELU, 2 ReLU.
// One wave computes one 16x16 C tile.  Fragment layouts per CDNA5 ISA 7.12.2.
// ---------------------------------------------------------------------------
__global__ __launch_bounds__(256)
void k_gemm(const _Float16* __restrict__ A, const _Float16* __restrict__ B,
            const float* __restrict__ bias1, const float* __restrict__ bias2,
            float* __restrict__ C, _Float16* __restrict__ Ch, _Float16* __restrict__ ChT,
            int M, int N, int K,
            int lda, int ldb, int ldc, int ldch, int ldcht,
            long aB, long bB, long oB1, long oB2, int bdiv, int act)
{
  int mt=(M+15)>>4, nt=(N+15)>>4;
  int tiles=mt*nt;
  int wave = blockIdx.x*(blockDim.x>>5) + (threadIdx.x>>5);
  if(wave>=tiles) return;                       // wave-uniform: EXEC stays all-ones
  int lane = threadIdx.x & 31;
  int tm = wave/nt, tn = wave - tm*nt;
  long y = blockIdx.y;
  const _Float16* Ab = A + y*aB;
  const _Float16* Bb = B + y*bB;
  long oOff = (y/(long)bdiv)*oB1 + (y%(long)bdiv)*oB2;
  int row = (tm<<4) + (lane&15);
  int col = (tn<<4) + (lane&15);
  int ka = (lane>>4)<<3;   // A: K-half select (0 / 8)
  int kb = (lane>>4)<<3;   // B: K-half select (0 / 8)
  v8f acc = {};
  for(int k0=0;k0<K;k0+=32){
    const _Float16* ap = Ab + (long)row*lda + k0 + ka;
    const _Float16* bp = Bb + (long)col*ldb + k0 + kb;
    v8h alo = *reinterpret_cast<const v8h*>(ap);
    v8h ahi = *reinterpret_cast<const v8h*>(ap+16);
    v8h blo = *reinterpret_cast<const v8h*>(bp);
    v8h bhi = *reinterpret_cast<const v8h*>(bp+16);
    if(k0+32<K){ __builtin_prefetch((const void*)(ap+32),0,3);
                 __builtin_prefetch((const void*)(bp+32),0,3); }
    v16h av, bv;
#pragma unroll
    for(int j=0;j<8;++j){ av[j]=alo[j]; av[j+8]=ahi[j]; bv[j]=blo[j]; bv[j+8]=bhi[j]; }
    acc = __builtin_amdgcn_wmma_f32_16x16x32_f16(false, av, false, bv,
                                                 (short)0, acc, false, false);
  }
  int cn = (tn<<4) + (lane&15);
  bool nok = cn < N;
  float bsum = 0.f;
  if(nok){ if(bias1) bsum += bias1[cn]; if(bias2) bsum += bias2[cn]; }
#pragma unroll
  for(int r=0;r<8;++r){
    int cm = (tm<<4) + r + ((lane>>4)<<3);      // C layout: VGPR r -> M=r(+8)
    if(nok && cm<M){
      float v = acc[r] + bsum;
      if(act==1)      v = v>0.f ? v : expf(v)-1.f;
      else if(act==2) v = fmaxf(v,0.f);
      if(C)   C  [oOff + (long)cm*ldc  + cn] = v;
      if(Ch)  Ch [oOff + (long)cm*ldch + cn] = (_Float16)v;
      if(ChT) ChT[oOff + (long)cn*ldcht+ cm] = (_Float16)v;
    }
  }
}

// ------------------------- utility kernels ---------------------------------
__global__ void k_zero(float* p, long n){
  long i = blockIdx.x*(long)blockDim.x+threadIdx.x; if(i<n) p[i]=0.f;
}
__global__ void k_cvt(const float* s, _Float16* d, long n){
  long i = blockIdx.x*(long)blockDim.x+threadIdx.x; if(i<n) d[i]=(_Float16)s[i];
}
// f32 [R,C] -> f16 [PR,PC] zero-padded
__global__ void k_cvt_pad(const float* __restrict__ s, _Float16* __restrict__ d,
                          int R,int C,int PR,int PC){
  long i = blockIdx.x*(long)blockDim.x+threadIdx.x;
  if(i >= (long)PR*PC) return;
  int r = (int)(i/PC), c = (int)(i - (long)r*PC);
  d[i] = (_Float16)((r<R && c<C) ? s[(long)r*C+c] : 0.f);
}
// f32 [R,C] -> f16 transposed [C,R]
__global__ void k_cvt_t(const float* __restrict__ s, _Float16* __restrict__ d,
                        int R,int C){
  long i = blockIdx.x*(long)blockDim.x+threadIdx.x;
  if(i >= (long)R*C) return;
  int r = (int)(i/C), c = (int)(i - (long)r*C);
  d[(long)c*R + r] = (_Float16)s[i];
}

// ------------------------- attention ---------------------------------------
__global__ void k_scores(const float* __restrict__ qkv, float* __restrict__ sc){
  int i = blockIdx.x*blockDim.x+threadIdx.x;           // [b,h,qi,j]
  if(i >= BS*AH*T*T) return;
  int j=i%T, qi=(i/T)%T, h=(i/(T*T))%AH, b=i/(T*T*AH);
  const float* q = qkv + (long)(b*T+qi)*384 + h*16;
  const float* k = qkv + (long)(b*T+j )*384 + 128 + h*16;
  float s=0.f;
  for(int d=0; d<16; ++d) s += q[d]*k[d];
  sc[i] = s*0.25f;                                     // 1/sqrt(16)
}
__global__ void k_softmax48(float* __restrict__ sc){
  int row = blockIdx.x*(blockDim.x>>5)+(threadIdx.x>>5);
  if(row >= BS*AH*T) return;
  int lane = threadIdx.x&31;
  float* r = sc + (long)row*T;
  float m=-3.4e38f;
  for(int j=lane;j<T;j+=32) m=fmaxf(m,r[j]);
  for(int o=16;o;o>>=1) m=fmaxf(m,__shfl_xor(m,o,32));
  float s=0.f;
  for(int j=lane;j<T;j+=32){ float e=expf(r[j]-m); r[j]=e; s+=e; }
  for(int o=16;o;o>>=1) s+=__shfl_xor(s,o,32);
  float inv=1.f/s;
  for(int j=lane;j<T;j+=32) r[j]*=inv;
}
__global__ void k_ctx(const float* __restrict__ sc, const float* __restrict__ qkv,
                      _Float16* __restrict__ ctxh){
  int i = blockIdx.x*blockDim.x+threadIdx.x;           // [b,h,qi,d]
  if(i >= BS*AH*T*16) return;
  int d=i&15, qi=(i>>4)%T, h=(i/(16*T))%AH, b=i/(16*T*AH);
  const float* srow = sc + ((long)(b*AH+h)*T+qi)*T;
  const float* vb = qkv + 256 + h*16 + d;
  float acc=0.f;
  for(int j=0;j<T;++j) acc += srow[j]*vb[(long)(b*T+j)*384];
  ctxh[(long)(b*T+qi)*Hh + h*16 + d] = (_Float16)acc;
}
// y = LN(a+b)*g+beta ; writes f32 + f16
__global__ void k_ln(const float* __restrict__ a, const float* __restrict__ b2,
                     const float* __restrict__ g, const float* __restrict__ be,
                     float* __restrict__ out, _Float16* __restrict__ outh){
  int row = blockIdx.x*(blockDim.x>>5)+(threadIdx.x>>5);
  if(row >= BT) return;
  int lane = threadIdx.x&31;
  const float* pa=a+(long)row*Hh; const float* pb=b2+(long)row*Hh;
  float x[4]; float s=0.f;
  for(int u=0;u<4;++u){ x[u]=pa[lane+32*u]+pb[lane+32*u]; s+=x[u]; }
  for(int o=16;o;o>>=1) s+=__shfl_xor(s,o,32);
  float mean=s*(1.f/Hh); float v=0.f;
  for(int u=0;u<4;++u){ float d=x[u]-mean; v+=d*d; }
  for(int o=16;o;o>>=1) v+=__shfl_xor(v,o,32);
  float rs=rsqrtf(v*(1.f/Hh)+1e-5f);
  for(int u=0;u<4;++u){ int c=lane+32*u;
    float yv=(x[u]-mean)*rs*g[c]+be[c];
    out[(long)row*Hh+c]=yv; outh[(long)row*Hh+c]=(_Float16)yv; }
}

// ------------------------- LSTMs (recurrent parts) --------------------------
__global__ void k_lstm1_step(const float* __restrict__ GX, const float* __restrict__ Whh,
                             const float* __restrict__ hin, const float* __restrict__ cin,
                             float* __restrict__ hout, float* __restrict__ cout,
                             float* __restrict__ YL, int t){
  int idx = blockIdx.x*blockDim.x+threadIdx.x;
  if(idx >= BS*INs) return;
  int b=idx/INs, j=idx-b*INs, p=b*T+t;
  const float* gr = GX + (long)p*G1NP;
  float gi=gr[j], gf=gr[INs+j], gg=gr[2*INs+j], go=gr[3*INs+j];
  const float* h  = hin + b*INs;
  const float* wi = Whh + (long)j*INs;
  const float* wf = Whh + (long)(INs+j)*INs;
  const float* wg = Whh + (long)(2*INs+j)*INs;
  const float* wo = Whh + (long)(3*INs+j)*INs;
  for(int k=0;k<INs;++k){ float hk=h[k]; gi+=hk*wi[k]; gf+=hk*wf[k]; gg+=hk*wg[k]; go+=hk*wo[k]; }
  float c = sigf(gf)*cin[idx] + sigf(gi)*tanhf(gg);
  float hh = sigf(go)*tanhf(c);
  cout[idx]=c; hout[idx]=hh;
  YL[(long)p*INs + j] = hh;
}
__global__ void k_lstm2_step(const float* __restrict__ GX, const float* __restrict__ Whh,
                             const float* __restrict__ hin, const float* __restrict__ cin,
                             float* __restrict__ hout, float* __restrict__ cout, int t){
  int idx = blockIdx.x*blockDim.x+threadIdx.x;
  if(idx >= BS*Hh) return;
  int b=idx/Hh, j=idx-b*Hh, p=b*T+t;
  const float* gr = GX + (long)p*512;
  float gi=gr[j], gf=gr[Hh+j], gg=gr[2*Hh+j], go=gr[3*Hh+j];
  const float* h  = hin + b*Hh;
  const float* wi = Whh + (long)j*Hh;
  const float* wf = Whh + (long)(Hh+j)*Hh;
  const float* wg = Whh + (long)(2*Hh+j)*Hh;
  const float* wo = Whh + (long)(3*Hh+j)*Hh;
  for(int k=0;k<Hh;++k){ float hk=h[k]; gi+=hk*wi[k]; gf+=hk*wf[k]; gg+=hk*wg[k]; go+=hk*wo[k]; }
  float c = sigf(gf)*cin[idx] + sigf(gi)*tanhf(gg);
  float hh = sigf(go)*tanhf(c);
  cout[idx]=c; hout[idx]=hh;
}

// ------------------------- GAT ----------------------------------------------
// WhT[pk][h][n] = Yg[p,n,:2] . W1[k,:,h]  (f16, padded n zeroed).
// Yg[p,n,c] = YL_flat[n*2*BT + p*2 + c]  (the reference's raw reshape)
__global__ void k_wht(const float* __restrict__ YL, const float* __restrict__ W1,
                      _Float16* __restrict__ WHT){
  long i = blockIdx.x*(long)blockDim.x+threadIdx.x;    // [pk][h][n]
  if(i >= (long)3072*Hh*NP) return;
  int n = (int)(i%NP); long r = i/NP; int h = (int)(r%Hh); int pk = (int)(r/Hh);
  int p = pk>>2, k = pk&3;
  float v = 0.f;
  if(n < Nn){
    float x0 = YL[(long)n*(2*BT) + p*2 + 0];
    float x1 = YL[(long)n*(2*BT) + p*2 + 1];
    v = x0*W1[k*256 + h] + x1*W1[k*256 + 128 + h];
  }
  WHT[i] = (_Float16)v;
}
__global__ void k_f12(const _Float16* __restrict__ WHT, const float* __restrict__ a1,
                      float* __restrict__ F1, float* __restrict__ F2){
  int i = blockIdx.x*blockDim.x+threadIdx.x;           // [pk][n]
  if(i >= 3072*NP) return;
  int n = i%NP, pk = i/NP, k = pk&3;
  float f1=0.f, f2=0.f;
  if(n < Nn){
    const _Float16* col = WHT + (long)pk*Hh*NP + n;
    for(int h=0;h<Hh;++h){ float w=(float)col[(long)h*NP];
      f1 += w*a1[k*256+h]; f2 += w*a1[k*256+128+h]; }
  }
  F1[i]=f1; F2[i]=f2;
}
// masked leaky-relu softmax row -> f16 att, padded cols/rows zeroed
__global__ void k_gat_soft(const float* __restrict__ F1, const float* __restrict__ F2,
                           const int* __restrict__ adj, _Float16* __restrict__ ATT,
                           int nrows){
  int row = blockIdx.x*(blockDim.x>>5)+(threadIdx.x>>5); // g*NP + i
  if(row >= nrows) return;
  int lane = threadIdx.x&31;
  int i = row%NP, g = row/NP;
  _Float16* out = ATT + (long)row*NP;
  if(i >= Nn){ for(int j=lane;j<NP;j+=32) out[j]=(_Float16)0.f; return; }
  float f1 = F1[(long)g*NP + i];
  float vals[5]; float m=-3.4e38f;
#pragma unroll
  for(int u=0;u<5;++u){
    int j = lane + 32*u; float v = -3.4e38f;
    if(j < Nn){
      float e = f1 + F2[(long)g*NP + j];
      e = (e>=0.f) ? e : 0.1f*e;                       // leaky_relu(0.1)
      v = (adj[i*Nn+j] > 0) ? e : -9e15f;
    }
    vals[u]=v; m=fmaxf(m,v);
  }
  for(int o=16;o;o>>=1) m=fmaxf(m,__shfl_xor(m,o,32));
  float s=0.f;
#pragma unroll
  for(int u=0;u<5;++u){
    int j = lane + 32*u;
    float e = (j<Nn) ? expf(vals[u]-m) : 0.f;
    vals[u]=e; s+=e;
  }
  for(int o=16;o;o>>=1) s+=__shfl_xor(s,o,32);
  float inv = 1.f/s;
#pragma unroll
  for(int u=0;u<5;++u){ int j=lane+32*u; out[j] = (_Float16)(vals[u]*inv); }
}
__global__ void k_g12(const _Float16* __restrict__ WHOT, const float* __restrict__ a2,
                      float* __restrict__ G1, float* __restrict__ G2){
  int i = blockIdx.x*blockDim.x+threadIdx.x;           // [p][n]
  if(i >= BT*NP) return;
  int n=i%NP, p=i/NP;
  float g1=0.f, g2=0.f;
  if(n < Nn){
    const _Float16* col = WHOT + (long)p*NP + n;       // WHOT[h][p*NP+n]
    for(int h=0;h<Hh;++h){ float w=(float)col[(long)h*(BT*NP)];
      g1 += w*a2[h]; g2 += w*a2[128+h]; }
  }
  G1[i]=g1; G2[i]=g2;
}
// LSTM2 input = GAT-out (raw (n,p,h)-ordered flat) + elu skip-connection
__global__ void k_z(const float* __restrict__ F, const float* __restrict__ X,
                    const float* __restrict__ Wsc, const float* __restrict__ bsc,
                    _Float16* __restrict__ Z){
  long i = blockIdx.x*(long)blockDim.x+threadIdx.x;    // [p][q], q in [0,18816)
  if(i >= (long)BT*NH) return;
  int q = (int)(i%NH); int p = (int)(i/NH);
  int n = q>>7, h = q&127;
  float fo = F[(long)n*(BT*Hh) + p*Hh + h];
  float x0 = X[(long)p*INs + 2*n], x1 = X[(long)p*INs + 2*n + 1];
  float s = x0*Wsc[h*2] + x1*Wsc[h*2+1] + bsc[h];
  s = (s>0.f) ? s : expf(s)-1.f;
  Z[i] = (_Float16)(fo + s);
}

// ===========================================================================
extern "C" void kernel_launch(void* const* d_in, const int* in_sizes, int n_in,
                              void* d_out, int out_size, void* d_ws, size_t ws_size,
                              hipStream_t stream) {
  const float* X    =(const float*)d_in[0];
  const int*   ADJ  =(const int*)  d_in[1];
  const float* WT   =(const float*)d_in[2];  const float* BT_  =(const float*)d_in[3];
  const float* EINW =(const float*)d_in[4];  const float* EINB =(const float*)d_in[5];
  const float* EOUTW=(const float*)d_in[6];  const float* EOUTB=(const float*)d_in[7];
  const float* LN1G =(const float*)d_in[8];  const float* LN1B =(const float*)d_in[9];
  const float* LN2G =(const float*)d_in[10]; const float* LN2B =(const float*)d_in[11];
  const float* FF1W =(const float*)d_in[12]; const float* FF1B =(const float*)d_in[13];
  const float* FF2W =(const float*)d_in[14]; const float* FF2B =(const float*)d_in[15];
  const float* WIH1 =(const float*)d_in[16]; const float* WHH1 =(const float*)d_in[17];
  const float* BIH1 =(const float*)d_in[18]; const float* BHH1 =(const float*)d_in[19];
  const float* GW1  =(const float*)d_in[20]; const float* GA1  =(const float*)d_in[21];
  const float* GW2  =(const float*)d_in[22]; const float* GA2  =(const float*)d_in[23];
  const float* WSC  =(const float*)d_in[24]; const float* BSC  =(const float*)d_in[25];
  const float* WIH2 =(const float*)d_in[26]; const float* WHH2 =(const float*)d_in[27];
  const float* BIH2 =(const float*)d_in[28]; const float* BHH2 =(const float*)d_in[29];
  const float* WFC  =(const float*)d_in[30]; const float* BFC  =(const float*)d_in[31];
  float* OUT = (float*)d_out;

  // bump allocator over d_ws
  char* wp = (char*)d_ws;
  auto alloc = [&](size_t bytes)->void*{ void* p=wp; wp += (bytes+255)&~(size_t)255; return p; };
  auto ah = [&](long n)->_Float16*{ return (_Float16*)alloc((size_t)n*2); };
  auto af = [&](long n)->float*   { return (float*)   alloc((size_t)n*4); };

  _Float16 *Xh   = ah((long)BT*INP),     *WTh  = ah((long)Hh*INP);
  _Float16 *EINh = ah(384*Hh),           *EOUTh= ah(Hh*Hh);
  _Float16 *FF1Wh= ah(2048*Hh),          *FF2Wh= ah((long)Hh*2048);
  _Float16 *WIH1h= ah((long)G1NP*Hh),    *W2Th = ah(Hh*512);
  _Float16 *WIH2h= ah((long)512*NH),     *WFCh = ah(304*Hh);
  _Float16 *Y0h  = ah((long)BT*Hh),      *CTXh = ah((long)BT*Hh);
  _Float16 *Y1h  = ah((long)BT*Hh),      *FF1h = ah((long)BT*2048);
  _Float16 *Y2h  = ah((long)BT*Hh);
  _Float16 *WHT  = ah((long)3072*Hh*NP);          // [pk][128][160]
  _Float16 *ATT1 = ah((long)3072*NP*NP);          // [pk][160][160]
  _Float16 *H1Ch = ah((long)BT*NP*512);           // [p][160][512] head-concat
  _Float16 *WHOTh= ah((long)Hh*BT*NP);            // [128][122880] = Who^T
  _Float16 *ATT2 = ah((long)BT*NP*NP);            // [p][160][160]
  _Float16 *Zh   = ah((long)BT*NH);               // LSTM2 input, f16
  _Float16 *H2h  = ah(BS*Hh);
  float *Y0 = af((long)BT*Hh),  *QKV = af((long)BT*384), *SCb = af((long)BS*AH*T*T);
  float *AO = af((long)BT*Hh),  *Y1  = af((long)BT*Hh),  *FF2 = af((long)BT*Hh);
  float *Y2 = af((long)BT*Hh),  *G1X = af((long)BT*G1NP);
  float *HC1= af(4L*BS*INs),    *YL  = af((long)BT*INs);
  float *F1b= af((long)3072*NP),*F2b = af((long)3072*NP);
  float *G1b= af((long)BT*NP),  *G2b = af((long)BT*NP);
  float *Fb = af((long)Nn*BT*Hh);                 // GAT out, raw (n,p,h) order
  float *G2X= af((long)BT*512), *HC2 = af(4L*BS*Hh);

  auto g1d = [](long n){ return dim3((unsigned)((n+255)/256)); };
  auto gemm = [&](const _Float16* A,const _Float16* B,const float* b1,const float* b2,
                  float* C,_Float16* Ch,_Float16* ChT,
                  int M,int N,int K,int lda,int ldb,int ldc,int ldch,int ldcht,
                  long aB,long bB,long oB1,long oB2,int bdiv,int batch,int act){
    int tiles = ((M+15)/16)*((N+15)/16);
    dim3 g((tiles+7)/8, batch);
    k_gemm<<<g,256,0,stream>>>(A,B,b1,b2,C,Ch,ChT,M,N,K,lda,ldb,ldc,ldch,ldcht,
                               aB,bB,oB1,oB2,bdiv,act);
  };

  // --- weight/input conversions (f32 -> f16, zero-padded) ---
  k_cvt_pad<<<g1d((long)BT*INP),256,0,stream>>>(X,   Xh,   BT,INs, BT,INP);
  k_cvt_pad<<<g1d((long)Hh*INP),256,0,stream>>>(WT,  WTh,  Hh,INs, Hh,INP);
  k_cvt_pad<<<g1d(384L*Hh),256,0,stream>>>(EINW, EINh, 384,Hh, 384,Hh);
  k_cvt_pad<<<g1d((long)Hh*Hh),256,0,stream>>>(EOUTW,EOUTh,Hh,Hh, Hh,Hh);
  k_cvt_pad<<<g1d(2048L*Hh),256,0,stream>>>(FF1W, FF1Wh,2048,Hh,2048,Hh);
  k_cvt_pad<<<g1d((long)Hh*2048),256,0,stream>>>(FF2W, FF2Wh,Hh,2048,Hh,2048);
  k_cvt_pad<<<g1d((long)G1NP*Hh),256,0,stream>>>(WIH1, WIH1h,G1N,Hh,G1NP,Hh);
  k_cvt_t  <<<g1d(512L*Hh),256,0,stream>>>(GW2, W2Th, 512, Hh);       // -> [128,512]
  k_cvt_pad<<<g1d(512L*NH),256,0,stream>>>(WIH2, WIH2h,512,NH,512,NH);
  k_cvt_pad<<<g1d(304L*Hh),256,0,stream>>>(WFC,  WFCh, OUTs,Hh,304,Hh);
  k_zero<<<g1d(4L*BS*INs),256,0,stream>>>(HC1, 4L*BS*INs);
  k_zero<<<g1d(4L*BS*Hh ),256,0,stream>>>(HC2, 4L*BS*Hh);

  // --- input projection: Y0 = elu(X @ Wt^T + b) ---
  gemm(Xh,WTh,BT_,nullptr, Y0,Y0h,nullptr, BT,Hh,INP, INP,INP,Hh,Hh,0,
       0,0,0,0,1,1, /*elu*/1);
  // --- MHA ---
  gemm(Y0h,EINh,EINB,nullptr, QKV,nullptr,nullptr, BT,384,Hh, Hh,Hh,384,0,0,
       0,0,0,0,1,1, 0);
  k_scores  <<<g1d((long)BS*AH*T*T),256,0,stream>>>(QKV,SCb);
  k_softmax48<<<dim3((BS*AH*T+7)/8),256,0,stream>>>(SCb);
  k_ctx     <<<g1d((long)BS*AH*T*16),256,0,stream>>>(SCb,QKV,CTXh);
  gemm(CTXh,EOUTh,EOUTB,nullptr, AO,nullptr,nullptr, BT,Hh,Hh, Hh,Hh,Hh,0,0,
       0,0,0,0,1,1, 0);
  k_ln<<<dim3(BT/8),256,0,stream>>>(Y0,AO,LN1G,LN1B,Y1,Y1h);
  // --- FFN ---
  gemm(Y1h,FF1Wh,FF1B,nullptr, nullptr,FF1h,nullptr, BT,2048,Hh, Hh,Hh,0,2048,0,
       0,0,0,0,1,1, /*relu*/2);
  gemm(FF1h,FF2Wh,FF2B,nullptr, FF2,nullptr,nullptr, BT,Hh,2048, 2048,2048,Hh,0,0,
       0,0,0,0,1,1, 0);
  k_ln<<<dim3(BT/8),256,0,stream>>>(Y1,FF2,LN2G,LN2B,Y2,Y2h);
  // --- LSTM1: batched input projection + 48 recurrent steps ---
  gemm(Y2h,WIH1h,BIH1,BHH1, G1X,nullptr,nullptr, BT,G1N,Hh, Hh,Hh,G1NP,0,0,
       0,0,0,0,1,1, 0);
  {
    float *h0=HC1, *h1=HC1+BS*INs, *c0=HC1+2L*BS*INs, *c1=HC1+3L*BS*INs;
    for(int t=0;t<T;++t){
      const float* hin=(t&1)?h1:h0; float* hout=(t&1)?h0:h1;
      const float* cin=(t&1)?c1:c0; float* cout=(t&1)?c0:c1;
      k_lstm1_step<<<g1d((long)BS*INs),256,0,stream>>>(G1X,WHH1,hin,cin,hout,cout,YL,t);
    }
  }
  // --- GAT layer 1 ---
  k_wht<<<g1d((long)3072*Hh*NP),256,0,stream>>>(YL,GW1,WHT);
  k_f12<<<g1d((long)3072*NP),256,0,stream>>>(WHT,GA1,F1b,F2b);
  k_gat_soft<<<dim3((3072*NP+7)/8),256,0,stream>>>(F1b,F2b,ADJ,ATT1,3072*NP);
  // h1 = elu(att @ Wh): batch pk=3072, Ch -> head-concat [p][160][k*128+h]
  gemm(ATT1,WHT,nullptr,nullptr, nullptr,H1Ch,nullptr, NP,Hh,NP,
       NP,NP,0,512,0, (long)NP*NP,(long)Hh*NP, (long)NP*512,128, 4,3072, /*elu*/1);
  // --- GAT layer 2 ---
  // Who = h1c @ W2 (flattened M=BT*NP), store transposed f16 [128][BT*NP]
  gemm(H1Ch,W2Th,nullptr,nullptr, nullptr,nullptr,WHOTh, BT*NP,Hh,512,
       512,512,0,0,BT*NP, 0,0,0,0,1,1, 0);
  k_g12<<<g1d((long)BT*NP),256,0,stream>>>(WHOTh,GA2,G1b,G2b);
  k_gat_soft<<<dim3((BT*NP+7)/8),256,0,stream>>>(G1b,G2b,ADJ,ATT2,BT*NP);
  // out = elu(att2 @ Who): write straight into (n,p,h)-ordered flat buffer
  gemm(ATT2,WHOTh,nullptr,nullptr, Fb,nullptr,nullptr, Nn,Hh,NP,
       NP,BT*NP,BT*Hh,0,0, (long)NP*NP,(long)NP, (long)Hh,0, 1,BT, /*elu*/1);
  // --- skip connection + LSTM2 ---
  k_z<<<g1d((long)BT*NH),256,0,stream>>>(Fb,X,WSC,BSC,Zh);
  gemm(Zh,WIH2h,BIH2,BHH2, G2X,nullptr,nullptr, BT,512,NH, NH,NH,512,0,0,
       0,0,0,0,1,1, 0);
  {
    float *h0=HC2, *h1=HC2+BS*Hh, *c0=HC2+2L*BS*Hh, *c1=HC2+3L*BS*Hh;
    for(int t=0;t<T;++t){
      const float* hin=(t&1)?h1:h0; float* hout=(t&1)?h0:h1;
      const float* cin=(t&1)?c1:c0; float* cout=(t&1)?c0:c1;
      k_lstm2_step<<<g1d((long)BS*Hh),256,0,stream>>>(G2X,WHH2,hin,cin,hout,cout,t);
    }
  }
  // final h lives in HC2[0] (48 steps, even count)
  k_cvt<<<g1d((long)BS*Hh),256,0,stream>>>(HC2,H2h,(long)BS*Hh);
  gemm(H2h,WFCh,BFC,nullptr, OUT,nullptr,nullptr, BS,OUTs,Hh, Hh,Hh,OUTs,0,0,
       0,0,0,0,1,1, 0);
}
